// HierachicalRNN_7945689497866
// MI455X (gfx1250) — matrix operand order repaired and checked by reading, not compiled, not run on hardware
//
#include <hip/hip_runtime.h>
#include <hip/hip_bf16.h>

typedef __attribute__((ext_vector_type(16))) _Float16 v16h;
typedef __attribute__((ext_vector_type(8)))  _Float16 v8h;
typedef __attribute__((ext_vector_type(4)))  _Float16 v4h;
typedef __attribute__((ext_vector_type(8)))  float    v8f;

union V16 { v16h v; v8h h[2]; };

// Async load-to-LDS path (CDNA5). Guarded: falls back to direct global reads.
#if defined(__has_builtin)
#  if __has_builtin(__builtin_amdgcn_global_load_async_to_lds_b128)
#    define USE_ASYNC 1
#  endif
#endif
#ifndef USE_ASYNC
#  define USE_ASYNC 0
#endif

#if USE_ASYNC
#  if __has_builtin(__builtin_amdgcn_s_wait_asynccnt)
#    define WAIT_ASYNC() __builtin_amdgcn_s_wait_asynccnt(0)
#  else
#    define WAIT_ASYNC() asm volatile("s_wait_asynccnt 0x0" ::: "memory")
#  endif
// Builtin signature (from compiler diagnostic): params are pointers to
// 4 x int vectors; src in global (as1), dst in LDS (as3).
typedef int vi4 __attribute__((vector_size(16)));
typedef __attribute__((address_space(1))) vi4 gvi4;
typedef __attribute__((address_space(3))) vi4 lvi4;
#endif

// ---------------------------------------------------------------------------
// f32 -> f16 conversion (weights, embed table)
__global__ void cvt_f32_f16(const float* __restrict__ s, _Float16* __restrict__ d, size_t n) {
    size_t i = (size_t)blockIdx.x * blockDim.x + threadIdx.x;
    if (i < n) d[i] = (_Float16)s[i];
}

__global__ void copy_f32(const float* __restrict__ s, float* __restrict__ d, int n) {
    int i = blockIdx.x * blockDim.x + threadIdx.x;
    if (i < n) d[i] = s[i];
}

// ---------------------------------------------------------------------------
// Embedding gather: rows ordered [(h,l), w, b] to match GI layout.
// tokens: [B, NH, NL, NW] ; Eh: [V, 256] f16 ; X: [NH*NL*NW*B, 256] f16
__global__ void gather_embed(const int* __restrict__ tok,
                             const _Float16* __restrict__ Eh,
                             _Float16* __restrict__ X) {
    int row = blockIdx.x;                 // 131072 rows
    int b = row & 31;
    int w = (row >> 5) & 31;
    int l = (row >> 10) & 15;
    int h = row >> 14;
    int t = tok[((b * 8 + h) * 16 + l) * 32 + w];
    const v4h* src = (const v4h*)(Eh + (size_t)t * 256);
    v4h* dst = (v4h*)(X + (size_t)row * 256);
    dst[threadIdx.x] = src[threadIdx.x];  // 64 threads x 4 halfs = 256
}

// ---------------------------------------------------------------------------
// Generic WMMA GEMM:  C[M,N] = A[M,K](f16,row) @ W[N,K]^T(f16,row) + bias
// One wave per (16 x 64) output tile: A tile reused across 4 N-tiles.
__global__ void gemm_wmma(const _Float16* __restrict__ A, int lda,
                          const _Float16* __restrict__ W,
                          const float* __restrict__ bias,
                          float* __restrict__ C, _Float16* __restrict__ Ch,
                          int ldc, int M, int N, int K) {
    int ntn4 = N >> 6;
    int tn4  = blockIdx.x % ntn4;
    int tm   = blockIdx.x / ntn4;
    int lane = threadIdx.x & 31;
    int r    = lane & 15;
    int sel  = (lane >> 4) & 1;

    const _Float16* Arow = A + (size_t)(tm * 16 + r) * lda;
    v8f acc0 = {}, acc1 = {}, acc2 = {}, acc3 = {};

    for (int k0 = 0; k0 < K; k0 += 32) {
        V16 a;
        a.h[0] = *(const v8h*)(Arow + k0 + sel * 8);
        a.h[1] = *(const v8h*)(Arow + k0 + sel * 8 + 16);
#pragma unroll
        for (int jn = 0; jn < 4; ++jn) {
            const _Float16* Wrow = W + (size_t)(tn4 * 64 + jn * 16 + r) * K;
            V16 b;
            b.h[0] = *(const v8h*)(Wrow + k0 + sel * 8);
            b.h[1] = *(const v8h*)(Wrow + k0 + sel * 8 + 16);
            v8f* acc = (jn == 0) ? &acc0 : (jn == 1) ? &acc1 : (jn == 2) ? &acc2 : &acc3;
            *acc = __builtin_amdgcn_wmma_f32_16x16x32_f16(
                false, a.v, false, b.v, (short)0, *acc, false, false);
        }
    }
#pragma unroll
    for (int jn = 0; jn < 4; ++jn) {
        v8f* acc = (jn == 0) ? &acc0 : (jn == 1) ? &acc1 : (jn == 2) ? &acc2 : &acc3;
        int n = (tn4 * 4 + jn) * 16 + r;
        float bn = bias ? bias[n] : 0.f;
#pragma unroll
        for (int j = 0; j < 8; ++j) {
            int m = tm * 16 + sel * 8 + j;
            float v = (*acc)[j] + bn;
            C[(size_t)m * ldc + n] = v;
            if (Ch) Ch[(size_t)m * ldc + n] = (_Float16)v;
        }
    }
}

// ---------------------------------------------------------------------------
// Bidirectional GRU scan with LDS-resident hidden state.
// One block = one (sequence, direction) chain. 8 waves, gh via WMMA from LDS.
// The next timestep's GI slice (32x384 f32, this direction only) is
// prefetched into a double-buffered LDS region with async load-to-LDS so the
// global fetch overlaps the current step's WMMA + gate math (serial chain).
// GI : [nseq, T, B, 768] (fwd gates cols 0..383, bwd 384..767)
// Whh: [768, 128] f16 ; bhh: [768] f32 ; h0: [2, B, 128] f32
// OUT: [nseq, T, B, 256] f32 (+f16 mirror), fwd cols 0..127, bwd 128..255
__global__ __launch_bounds__(256) void gru_wmma(const float* __restrict__ GI,
                         const _Float16* __restrict__ Whh,
                         const float* __restrict__ bhh,
                         const float* __restrict__ h0,
                         float* __restrict__ OUT,
                         _Float16* __restrict__ OUTh,
                         int T) {
    __shared__ __align__(16) float    hs[32][128];
    __shared__ __align__(16) _Float16 hh[32][128];
    __shared__ __align__(16) float    gh[32][384];
#if USE_ASYNC
    __shared__ __align__(16) float    gis[2][32][384];   // double-buffered GI slice
#endif

    int seq = blockIdx.x >> 1;
    int dir = blockIdx.x & 1;
    int tid = threadIdx.x;
    int lane = tid & 31, wave = tid >> 5;
    int r = lane & 15, sel = (lane >> 4) & 1;

#if USE_ASYNC
    auto issueSlice = [&](int tc, int buf) {
        const float* src = GI + ((size_t)seq * T + tc) * 32 * 768 + dir * 384;
        for (int c = tid; c < 32 * 96; c += 256) {          // 16B chunks
            int b = c / 96, o = (c % 96) * 4;               // o in floats
            const float* g = src + (size_t)b * 768 + o;
            float*       l = &gis[buf][b][o];
            __builtin_amdgcn_global_load_async_to_lds_b128(
                (gvi4*)(uintptr_t)g,
                (lvi4*)(uint32_t)(uintptr_t)l, 0, 0);
        }
    };
#endif

    for (int i = tid; i < 32 * 128; i += 256)
        hs[i >> 7][i & 127] = h0[dir * 32 * 128 + i];
#if USE_ASYNC
    issueSlice(dir ? T - 1 : 0, 0);
    WAIT_ASYNC();
#endif
    __syncthreads();

    for (int t = 0; t < T; ++t) {
        int tc = dir ? (T - 1 - t) : t;          // time-aligned (scan reverse)
        int cur = t & 1;
        (void)cur;
#if USE_ASYNC
        if (t + 1 < T) issueSlice(dir ? (T - 2 - t) : (t + 1), 1 - cur);
#endif
        for (int i = tid; i < 32 * 128; i += 256)
            hh[i >> 7][i & 127] = (_Float16)hs[i >> 7][i & 127];
        __syncthreads();

        // gh = h @ Whh[dir]^T + bhh : M=32, N=384, K=128 -> 48 tiles / 8 waves
        for (int tile = wave; tile < 48; tile += 8) {
            int tm = tile & 1, tn = tile >> 1;
            const _Float16* Ar = &hh[tm * 16 + r][0];
            const _Float16* Wr = Whh + (size_t)(dir * 384 + tn * 16 + r) * 128;
            v8f acc = {};
#pragma unroll
            for (int k0 = 0; k0 < 128; k0 += 32) {
                V16 a, b;
                a.h[0] = *(const v8h*)(Ar + k0 + sel * 8);      // ds_load_b128
                a.h[1] = *(const v8h*)(Ar + k0 + sel * 8 + 16);
                b.h[0] = *(const v8h*)(Wr + k0 + sel * 8);
                b.h[1] = *(const v8h*)(Wr + k0 + sel * 8 + 16);
                acc = __builtin_amdgcn_wmma_f32_16x16x32_f16(
                    false, a.v, false, b.v, (short)0, acc, false, false);
            }
            float bn = bhh[dir * 384 + tn * 16 + r];
#pragma unroll
            for (int j = 0; j < 8; ++j)
                gh[tm * 16 + sel * 8 + j][tn * 16 + r] = acc[j] + bn;
        }
        __syncthreads();

#if USE_ASYNC
        const float* gip = &gis[cur][0][0];
        const int gst = 384, gof = 0;
#else
        const float* gip = GI + ((size_t)seq * T + tc) * 32 * 768;
        const int gst = 768, gof = dir * 384;
#endif
        for (int i = tid; i < 32 * 128; i += 256) {
            int b = i >> 7, j = i & 127;
            float gir = gip[(size_t)b * gst + gof + j];
            float giz = gip[(size_t)b * gst + gof + 128 + j];
            float gin = gip[(size_t)b * gst + gof + 256 + j];
            float rr = 1.f / (1.f + __expf(-(gir + gh[b][j])));
            float zz = 1.f / (1.f + __expf(-(giz + gh[b][128 + j])));
            float nn = tanhf(gin + rr * gh[b][256 + j]);
            float hv = (1.f - zz) * nn + zz * hs[b][j];
            hs[b][j] = hv;
            size_t o = (((size_t)seq * T + tc) * 32 + b) * 256 + dir * 128 + j;
            OUT[o]  = hv;
            OUTh[o] = (_Float16)hv;
        }
#if USE_ASYNC
        WAIT_ASYNC();
#endif
        __syncthreads();
    }
}

// ---------------------------------------------------------------------------
// Batch-axis softmax + weighted sum over time.
// attn = softmax over B (lane axis) of LGT[t, :, f]; res = sum_t attn * OUT
__global__ void attn_reduce(const float* __restrict__ LGT,
                            const float* __restrict__ OUT,
                            float* __restrict__ RES,
                            _Float16* __restrict__ RESh,
                            int T) {
    int seq = blockIdx.x;
    int f   = blockIdx.y;      // 0..255
    int b   = threadIdx.x;     // 0..31 (one wave)
    float acc = 0.f;
    for (int t = 0; t < T; ++t) {
        size_t o = (((size_t)seq * T + t) * 32 + b) * 256 + f;
        float x = LGT[o];
        float mx = x;
        for (int m = 16; m >= 1; m >>= 1) mx = fmaxf(mx, __shfl_xor(mx, m, 32));
        float e = __expf(x - mx);
        float s = e;
        for (int m = 16; m >= 1; m >>= 1) s += __shfl_xor(s, m, 32);
        acc += (e / s) * OUT[o];
    }
    size_t ro = ((size_t)seq * 32 + b) * 256 + f;
    RES[ro] = acc;
    if (RESh) RESh[ro] = (_Float16)acc;
}

// ---------------------------------------------------------------------------
extern "C" void kernel_launch(void* const* d_in, const int* in_sizes, int n_in,
                              void* d_out, int out_size, void* d_ws, size_t ws_size,
                              hipStream_t stream) {
    (void)in_sizes; (void)n_in; (void)out_size; (void)ws_size;

    const float* h_hunk = (const float*)d_in[2];
    const float* h_sent = (const float*)d_in[3];
    const float* h_word = (const float*)d_in[4];
    const float* embedf = (const float*)d_in[5];

    char* cur = (char*)d_ws;
    auto alloc = [&](size_t bytes) -> void* {
        void* r = (void*)cur;
        cur += (bytes + 255) & ~(size_t)255;
        return r;
    };
    auto cvt = [&](const void* s, void* d, size_t n) {
        cvt_f32_f16<<<(unsigned)((n + 255) / 256), 256, 0, stream>>>((const float*)s, (_Float16*)d, n);
    };
    auto cpy = [&](const void* s, void* d, int n) {
        copy_f32<<<(unsigned)((n + 255) / 256), 256, 0, stream>>>((const float*)s, (float*)d, n);
    };
    auto gemm = [&](const _Float16* A, int lda, const _Float16* W, const float* bias,
                    float* C, _Float16* Ch, int ldc, int M, int N, int K) {
        gemm_wmma<<<(unsigned)((M / 16) * (N / 64)), 32, 0, stream>>>(
            A, lda, W, bias, C, Ch, ldc, M, N, K);
    };

    // ---- per-level weight prep (f16 fwd|bwd concatenation) -----------------
    struct LW { _Float16 *Wih, *Whh, *Wattn, *Wcomb; float *bih, *bhh; const float* battn; };
    auto prep = [&](int base) -> LW {
        LW w;
        w.Wih   = (_Float16*)alloc(768 * 256 * 2);
        w.Whh   = (_Float16*)alloc(768 * 128 * 2);
        w.bih   = (float*)alloc(768 * 4);
        w.bhh   = (float*)alloc(768 * 4);
        w.Wattn = (_Float16*)alloc(256 * 256 * 2);
        w.Wcomb = (_Float16*)alloc(256 * 256 * 2);
        cvt(d_in[base + 0], w.Wih,               384 * 256);   // fwd Wih
        cvt(d_in[base + 4], w.Wih + 384 * 256,   384 * 256);   // bwd Wih
        cvt(d_in[base + 1], w.Whh,               384 * 128);   // fwd Whh
        cvt(d_in[base + 5], w.Whh + 384 * 128,   384 * 128);   // bwd Whh
        cpy(d_in[base + 2], w.bih,       384);
        cpy(d_in[base + 6], w.bih + 384, 384);
        cpy(d_in[base + 3], w.bhh,       384);
        cpy(d_in[base + 7], w.bhh + 384, 384);
        cvt(d_in[base + 8],  w.Wattn, 256 * 256);
        cvt(d_in[base + 10], w.Wcomb, 256 * 256);
        w.battn = (const float*)d_in[base + 9];
        return w;
    };
    LW wW = prep(6), wS = prep(17), wH = prep(28);

    // ---- scratch buffers ---------------------------------------------------
    _Float16* embedh = (_Float16*)alloc((size_t)32000 * 256 * 2);
    _Float16* Xw     = (_Float16*)alloc((size_t)131072 * 256 * 2);
    char*     GIwB   = (char*)alloc((size_t)131072 * 768 * 4);      // reused for ANN/LGT
    float*    GIw    = (float*)GIwB;
    float*    OUTw   = (float*)alloc((size_t)131072 * 256 * 4);
    _Float16* OUTwh  = (_Float16*)alloc((size_t)131072 * 256 * 2);
    float*    sents  = (float*)alloc((size_t)4096 * 256 * 4);
    _Float16* sentsh = (_Float16*)alloc((size_t)4096 * 256 * 2);
    char*     GIsB   = (char*)alloc((size_t)4096 * 768 * 4);
    float*    GIs    = (float*)GIsB;
    float*    OUTs   = (float*)alloc((size_t)4096 * 256 * 4);
    _Float16* OUTsh  = (_Float16*)alloc((size_t)4096 * 256 * 2);
    float*    hunks  = (float*)alloc((size_t)256 * 256 * 4);
    _Float16* hunksh = (_Float16*)alloc((size_t)256 * 256 * 2);
    char*     GIhB   = (char*)alloc((size_t)256 * 768 * 4);
    float*    GIh    = (float*)GIhB;
    float*    OUTk   = (float*)alloc((size_t)256 * 256 * 4);
    _Float16* OUTkh  = (_Float16*)alloc((size_t)256 * 256 * 2);

    // ======================= WORD LEVEL ====================================
    cvt(embedf, embedh, (size_t)32000 * 256);
    gather_embed<<<131072, 64, 0, stream>>>((const int*)d_in[0], embedh, Xw);

    // GI = X @ Wih_cat^T + bih_cat : [131072, 768]
    gemm(Xw, 256, wW.Wih, wW.bih, GIw, nullptr, 768, 131072, 768, 256);
    // 256 chains x 32 steps
    gru_wmma<<<256, 256, 0, stream>>>(GIw, wW.Whh, wW.bhh, h_word, OUTw, OUTwh, 32);
    // attention (reuse GI region: ANN f32 | ANN f16 | LGT f32)
    {
        float*    ANN  = (float*)GIwB;
        _Float16* ANNh = (_Float16*)(GIwB + (size_t)131072 * 256 * 4);
        float*    LGT  = (float*)(GIwB + (size_t)131072 * 256 * 6);
        gemm(OUTwh, 256, wW.Wattn, wW.battn, ANN, ANNh, 256, 131072, 256, 256);
        gemm(ANNh, 256, wW.Wcomb, nullptr, LGT, nullptr, 256, 131072, 256, 256);
        attn_reduce<<<dim3(128, 256), 32, 0, stream>>>(LGT, OUTw, sents, sentsh, 32);
    }

    // ======================= SENT LEVEL ====================================
    gemm(sentsh, 256, wS.Wih, wS.bih, GIs, nullptr, 768, 4096, 768, 256);
    gru_wmma<<<16, 256, 0, stream>>>(GIs, wS.Whh, wS.bhh, h_sent, OUTs, OUTsh, 16);
    {
        float*    ANN  = (float*)GIsB;
        _Float16* ANNh = (_Float16*)(GIsB + (size_t)4096 * 256 * 4);
        float*    LGT  = (float*)(GIsB + (size_t)4096 * 256 * 6);
        gemm(OUTsh, 256, wS.Wattn, wS.battn, ANN, ANNh, 256, 4096, 256, 256);
        gemm(ANNh, 256, wS.Wcomb, nullptr, LGT, nullptr, 256, 4096, 256, 256);
        attn_reduce<<<dim3(8, 256), 32, 0, stream>>>(LGT, OUTs, hunks, hunksh, 16);
    }

    // ======================= HUNK LEVEL ====================================
    gemm(hunksh, 256, wH.Wih, wH.bih, GIh, nullptr, 768, 256, 768, 256);
    gru_wmma<<<2, 256, 0, stream>>>(GIh, wH.Whh, wH.bhh, h_hunk, OUTk, OUTkh, 8);
    {
        float*    ANN  = (float*)GIhB;
        _Float16* ANNh = (_Float16*)(GIhB + (size_t)256 * 256 * 4);
        float*    LGT  = (float*)(GIhB + (size_t)256 * 256 * 6);
        gemm(OUTkh, 256, wH.Wattn, wH.battn, ANN, ANNh, 256, 256, 256, 256);
        gemm(ANNh, 256, wH.Wcomb, nullptr, LGT, nullptr, 256, 256, 256, 256);
        attn_reduce<<<dim3(1, 256), 32, 0, stream>>>(LGT, OUTk, (float*)d_out, nullptr, 8);
    }
}
